// DeepSeekBlock_85624468013212
// MI455X (gfx1250) — compile-verified, matrix operand order checked by reading, not run on hardware
//
#include <hip/hip_runtime.h>
#include <hip/hip_bf16.h>

#define TT 1024
#define HH 16

typedef __bf16 bf16;
typedef __attribute__((ext_vector_type(8)))  __bf16 bf16x8;
typedef __attribute__((ext_vector_type(16))) __bf16 bf16x16;
typedef __attribute__((ext_vector_type(8)))  float  f32x8;
typedef int v4i __attribute__((vector_size(16)));
typedef __attribute__((address_space(1))) v4i* v4i_glb;
typedef __attribute__((address_space(3))) v4i* v4i_lds;

#if __has_builtin(__builtin_amdgcn_global_load_async_to_lds_b128)
#define HAS_ASYNC 1
#else
#define HAS_ASYNC 0
#endif

union BF16Frag { bf16x16 v16; bf16x8 v8[2]; };

__device__ __forceinline__ f32x8 wmma_bf16(bf16x16 a, bf16x16 b, f32x8 c) {
  // D = A(16x32) * B(32x16) + C, f32 accumulate
  return __builtin_amdgcn_wmma_f32_16x16x32_bf16(false, a, false, b, (short)0, c,
                                                 false, false);
}

__device__ __forceinline__ bf16x16 ldfrag(const bf16* lo, const bf16* hi) {
  BF16Frag f;
  f.v8[0] = *(const bf16x8*)lo;
  f.v8[1] = *(const bf16x8*)hi;
  return f.v16;
}

// 16-byte global -> LDS copy. CDNA5: async DMA, tracked by ASYNCcnt, no VGPRs.
__device__ __forceinline__ void cp16(void* lds_dst, const void* glb_src) {
#if HAS_ASYNC
  v4i* gs = (v4i*)glb_src;   // generic, const stripped
  v4i* ls = (v4i*)lds_dst;
  __builtin_amdgcn_global_load_async_to_lds_b128((v4i_glb)gs, (v4i_lds)ls, 0, 0);
#else
  *(uint4*)lds_dst = *(const uint4*)glb_src;
#endif
}

__device__ __forceinline__ void wait_async() {
#if HAS_ASYNC
# if __has_builtin(__builtin_amdgcn_s_wait_asynccnt)
  __builtin_amdgcn_s_wait_asynccnt(0);
# else
  asm volatile("s_wait_asynccnt 0" ::: "memory");
# endif
#endif
}

// ---------------------------------------------------------------------------
// GEMM: C[M,N] (f32) = A[M,K] (bf16, row-major) * B[N,K]^T (f32 weight (O,I))
// 256 threads = 8 waves (2 in M x 4 in N). Macro tile 128x128, K step 32.
// Double-buffered LDS; A staged with async-to-LDS, B staged f32->bf16.
// ---------------------------------------------------------------------------
__global__ __launch_bounds__(256)
void k_gemm(const bf16* __restrict__ A, const float* __restrict__ Bw,
            float* __restrict__ C, int M, int N, int K) {
  __shared__ bf16 As[2][128][32];
  __shared__ bf16 Bs[2][128][32];
  const int bm = blockIdx.y * 128;
  const int bn = blockIdx.x * 128;
  const int tid  = threadIdx.x;
  const int lane = tid & 31;
  const int wave = tid >> 5;
  const int wm = (wave >> 2) * 64;
  const int wn = (wave & 3) * 32;
  const int l16 = lane & 15;
  const int hf  = lane >> 4;
  const int r = tid >> 1;           // staging row
  const int c = (tid & 1) * 16;     // staging 16-elem chunk

  auto stageA = [&](int k0, int buf) {
    const bf16* src = A + (size_t)(bm + r) * K + k0 + c;
    cp16(&As[buf][r][c],     src);
    cp16(&As[buf][r][c + 8], src + 8);
  };
  auto stageB = [&](int k0, int buf) {
    const int nrow = bn + r;
    bf16 tmp[16];
    if (nrow < N) {
      const float4* ws = (const float4*)(Bw + (size_t)nrow * K + k0 + c);
#pragma unroll
      for (int q = 0; q < 4; ++q) {
        float4 v = ws[q];
        tmp[q*4+0] = (bf16)v.x; tmp[q*4+1] = (bf16)v.y;
        tmp[q*4+2] = (bf16)v.z; tmp[q*4+3] = (bf16)v.w;
      }
      if (k0 + 32 < K)
        __builtin_prefetch(Bw + (size_t)nrow * K + k0 + 32 + c, 0, 1);
    } else {
#pragma unroll
      for (int q = 0; q < 16; ++q) tmp[q] = (bf16)0.0f;
    }
#pragma unroll
    for (int q = 0; q < 16; ++q) Bs[buf][r][c + q] = tmp[q];
  };

  f32x8 acc[4][2] = {};

  stageA(0, 0);
  stageB(0, 0);
  wait_async();
  __syncthreads();

  const int nk = K >> 5;
  for (int kt = 0; kt < nk; ++kt) {
    const int buf = kt & 1;
    if (kt + 1 < nk) {              // overlap next tile's loads with compute
      stageA((kt + 1) * 32, buf ^ 1);
      stageB((kt + 1) * 32, buf ^ 1);
    }

    bf16x16 af[4];
    bf16x16 bfr[2];
#pragma unroll
    for (int i = 0; i < 4; ++i) {
      const bf16* row = &As[buf][wm + i*16 + l16][0];
      af[i] = ldfrag(row + hf*8, row + 16 + hf*8);
    }
#pragma unroll
    for (int j = 0; j < 2; ++j) {
      const bf16* row = &Bs[buf][wn + j*16 + l16][0];
      bfr[j] = ldfrag(row + hf*8, row + 16 + hf*8);
    }
#pragma unroll
    for (int i = 0; i < 4; ++i)
#pragma unroll
      for (int j = 0; j < 2; ++j)
        acc[i][j] = wmma_bf16(af[i], bfr[j], acc[i][j]);

    wait_async();
    __syncthreads();
  }

  // C layout: vgpr e -> row hf*8+e, col l16 within each 16x16 tile
#pragma unroll
  for (int i = 0; i < 4; ++i)
#pragma unroll
    for (int j = 0; j < 2; ++j) {
      const int col = bn + wn + j*16 + l16;
      if (col < N) {
#pragma unroll
        for (int e = 0; e < 8; ++e) {
          const int row = bm + wm + i*16 + hf*8 + e;
          C[(size_t)row * N + col] = acc[i][j][e];
        }
      }
    }
}

// ---------------------------------------------------------------------------
// RMSNorm over rows: out_bf16[row, 0..C) = in[row*stride + c] * rsqrt(ms+eps)*w
// ---------------------------------------------------------------------------
__global__ __launch_bounds__(256)
void k_rmsnorm(const float* __restrict__ in, const float* __restrict__ w,
               bf16* __restrict__ out, int C, int in_stride) {
  const int row = blockIdx.x;
  const float* x = in + (size_t)row * in_stride;
  float xv[8];
  float ss = 0.f;
  int cnt = 0;
  for (int c = threadIdx.x; c < C; c += 256) {
    float v = x[c];
    xv[cnt++] = v;
    ss += v * v;
  }
#pragma unroll
  for (int off = 16; off > 0; off >>= 1) ss += __shfl_xor(ss, off, 32);
  __shared__ float red[8];
  if ((threadIdx.x & 31) == 0) red[threadIdx.x >> 5] = ss;
  __syncthreads();
  float tot = 0.f;
#pragma unroll
  for (int i = 0; i < 8; ++i) tot += red[i];
  const float scale = rsqrtf(tot / (float)C + 1e-6f);
  cnt = 0;
  for (int c = threadIdx.x; c < C; c += 256)
    out[(size_t)row * C + c] = (bf16)(xv[cnt++] * scale * w[c]);
}

// ---------------------------------------------------------------------------
// Pack q -> qf (bf16) with RoPE on dims [128,192). One thread per element pair.
// ---------------------------------------------------------------------------
__global__ __launch_bounds__(256)
void k_pack_qf(const float* __restrict__ q, const float* __restrict__ fcos,
               const float* __restrict__ fsin, bf16* __restrict__ qf, int total) {
  const int idx = blockIdx.x * 256 + threadIdx.x;
  if (idx >= total) return;
  const int j   = idx % 96;
  const int th  = idx / 96;
  const int h   = th % HH;
  const int tok = th / HH;
  const int t   = tok & (TT - 1);
  const float* src = q  + ((size_t)tok * HH + h) * 192;
  bf16*        dst = qf + ((size_t)tok * HH + h) * 192;
  if (j < 64) {
    dst[2*j]   = (bf16)src[2*j];
    dst[2*j+1] = (bf16)src[2*j+1];
  } else {
    const int p = j - 64;
    const float c = fcos[t*32 + p], s = fsin[t*32 + p];
    const float x0 = src[128 + 2*p], x1 = src[128 + 2*p + 1];
    dst[128 + 2*p]     = (bf16)(x0*c - x1*s);
    dst[128 + 2*p + 1] = (bf16)(x0*s + x1*c);
  }
}

// ---------------------------------------------------------------------------
// Pack kv -> kf (k_nope | roped k_pe broadcast) bf16 and v -> vt transposed
// vt layout: [b, h, d(128), t(1024)] so attention B-fragments are contiguous.
// ---------------------------------------------------------------------------
__global__ __launch_bounds__(256)
void k_pack_kfv(const float* __restrict__ kv, const float* __restrict__ kvall,
                const float* __restrict__ fcos, const float* __restrict__ fsin,
                bf16* __restrict__ kf, bf16* __restrict__ vt, int total) {
  const int idx = blockIdx.x * 256 + threadIdx.x;
  if (idx >= total) return;
  const int j   = idx % 160;
  const int th  = idx / 160;
  const int h   = th % HH;
  const int tok = th / HH;
  const int t   = tok & (TT - 1);
  const int b   = tok >> 10;
  if (j < 64) {                       // k_nope pair
    const float* src = kv + ((size_t)tok * HH + h) * 256;
    bf16*        dst = kf + ((size_t)tok * HH + h) * 192;
    dst[2*j]   = (bf16)src[2*j];
    dst[2*j+1] = (bf16)src[2*j+1];
  } else if (j < 96) {                // roped k_pe pair (broadcast over heads)
    const int p = j - 64;
    const float c = fcos[t*32 + p], s = fsin[t*32 + p];
    const float* pe = kvall + (size_t)tok * 576 + 512;
    const float x0 = pe[2*p], x1 = pe[2*p + 1];
    bf16* dst = kf + ((size_t)tok * HH + h) * 192 + 128;
    dst[2*p]     = (bf16)(x0*c - x1*s);
    dst[2*p + 1] = (bf16)(x0*s + x1*c);
  } else {                            // v transpose pair
    const int dp = (j - 96) * 2;
    const float* src = kv + ((size_t)tok * HH + h) * 256 + 128;
    bf16* dst = vt + ((size_t)(b*HH + h) * 128) * (size_t)TT;
    dst[(size_t)dp * TT + t]       = (bf16)src[dp];
    dst[(size_t)(dp + 1) * TT + t] = (bf16)src[dp + 1];
  }
}

// ---------------------------------------------------------------------------
// Flash attention with WMMA. grid (T/64, H, B), 128 threads = 4 waves.
// Each wave owns 16 query rows. S = Q K^T (qk-dim 192), causal online softmax,
// O += P V. K/V tiles double-buffered via async-to-LDS.
// ---------------------------------------------------------------------------
__global__ __launch_bounds__(128)
void k_attn(const bf16* __restrict__ qf, const bf16* __restrict__ kf,
            const bf16* __restrict__ vt, bf16* __restrict__ y) {
  __shared__ bf16 Ks[2][64][192];   // 2 x 24 KB key tiles
  __shared__ bf16 Vs[2][128][64];   // 2 x 16 KB V tiles, [d][key]
  __shared__ bf16 Ps[4][16][64];    //  8 KB P round-trip (C-layout -> A-layout)
  const int qt = blockIdx.x, h = blockIdx.y, b = blockIdx.z;
  const int tid  = threadIdx.x;
  const int wave = tid >> 5, lane = tid & 31;
  const int l16 = lane & 15, hf = lane >> 4;

  auto stageKV = [&](int kt, int buf) {
    {   // K tile: 64 keys x 192
      const int r = tid >> 1, c0 = (tid & 1) * 96;
      const bf16* src = kf + ((size_t)(b * TT + kt * 64 + r) * HH + h) * 192 + c0;
      bf16* dst = &Ks[buf][r][c0];
#pragma unroll
      for (int i = 0; i < 12; ++i) cp16(dst + i * 8, src + i * 8);
    }
    {   // V tile transposed: 128 d x 64 keys
      const bf16* src =
          vt + ((size_t)(b*HH + h) * 128 + tid) * (size_t)TT + kt * 64;
      bf16* dst = &Vs[buf][tid][0];
#pragma unroll
      for (int i = 0; i < 8; ++i) cp16(dst + i * 8, src + i * 8);
    }
  };

  // Q fragments for this wave's 16 rows, 6 K-chunks of 32 (qk-dim 192)
  bf16x16 qfr[6];
  {
    const int tok = b * TT + qt * 64 + wave * 16 + l16;
    const bf16* qr = qf + ((size_t)tok * HH + h) * 192;
#pragma unroll
    for (int kc = 0; kc < 6; ++kc)
      qfr[kc] = ldfrag(qr + kc*32 + hf*8, qr + kc*32 + 16 + hf*8);
  }

  f32x8 oacc[8] = {};
  float mrow[8], lrow[8];
#pragma unroll
  for (int e = 0; e < 8; ++e) { mrow[e] = -3.0e38f; lrow[e] = 0.f; }

  stageKV(0, 0);
  wait_async();
  __syncthreads();

  for (int kt = 0; kt <= qt; ++kt) {
    const int buf = kt & 1;
    if (kt < qt) stageKV(kt + 1, buf ^ 1);   // overlap with compute

    // S = Q K^T : this wave's 16 rows x 64 keys
    f32x8 sacc[4] = {};
#pragma unroll
    for (int j = 0; j < 4; ++j) {
      const bf16* krow = &Ks[buf][j*16 + l16][0];
#pragma unroll
      for (int kc = 0; kc < 6; ++kc) {
        bf16x16 bfrag = ldfrag(krow + kc*32 + hf*8, krow + kc*32 + 16 + hf*8);
        sacc[j] = wmma_bf16(qfr[kc], bfrag, sacc[j]);
      }
    }

    // scale + causal mask (only diagonal tile needs masking)
    const float sc = 192.0f * -0.5f;
#pragma unroll
    for (int j = 0; j < 4; ++j)
#pragma unroll
      for (int e = 0; e < 8; ++e) {
        float v = sacc[j][e] * sc;
        if (kt == qt) {
          const int tcol = j*16 + l16;
          const int srow = wave*16 + hf*8 + e;
          if (tcol > srow) v -= 1.0e9f;
        }
        sacc[j][e] = v;
      }

    // online softmax: per-row max/sum across 4 tiles and 16 lanes
    float alpha[8];
#pragma unroll
    for (int e = 0; e < 8; ++e) {
      float v = fmaxf(fmaxf(sacc[0][e], sacc[1][e]), fmaxf(sacc[2][e], sacc[3][e]));
#pragma unroll
      for (int off = 8; off > 0; off >>= 1) v = fmaxf(v, __shfl_xor(v, off, 32));
      const float mnew = fmaxf(mrow[e], v);
      alpha[e] = __expf(mrow[e] - mnew);
      mrow[e] = mnew;
      lrow[e] *= alpha[e];
    }
#pragma unroll
    for (int n = 0; n < 8; ++n)
#pragma unroll
      for (int e = 0; e < 8; ++e) oacc[n][e] *= alpha[e];

#pragma unroll
    for (int j = 0; j < 4; ++j)
#pragma unroll
      for (int e = 0; e < 8; ++e) {
        const float pexp = __expf(sacc[j][e] - mrow[e]);
        sacc[j][e] = pexp;
        Ps[wave][hf*8 + e][j*16 + l16] = (bf16)pexp;   // C-layout -> LDS
      }
#pragma unroll
    for (int e = 0; e < 8; ++e) {
      float rs = sacc[0][e] + sacc[1][e] + sacc[2][e] + sacc[3][e];
#pragma unroll
      for (int off = 8; off > 0; off >>= 1) rs += __shfl_xor(rs, off, 32);
      lrow[e] += rs;
    }

    // O += P V : A = P (16x64, via LDS in A-layout), B = Vs columns
    bf16x16 pfr[2];
    {
      const bf16* prow = &Ps[wave][l16][0];
#pragma unroll
      for (int kc = 0; kc < 2; ++kc)
        pfr[kc] = ldfrag(prow + kc*32 + hf*8, prow + kc*32 + 16 + hf*8);
    }
#pragma unroll
    for (int n = 0; n < 8; ++n) {
      const bf16* vrow = &Vs[buf][n*16 + l16][0];
#pragma unroll
      for (int kc = 0; kc < 2; ++kc) {
        bf16x16 bfrag = ldfrag(vrow + kc*32 + hf*8, vrow + kc*32 + 16 + hf*8);
        oacc[n] = wmma_bf16(pfr[kc], bfrag, oacc[n]);
      }
    }
    wait_async();
    __syncthreads();
  }

  // epilogue: normalize and store y[token, h*128 + d] (bf16, A of wo-GEMM)
#pragma unroll
  for (int n = 0; n < 8; ++n)
#pragma unroll
    for (int e = 0; e < 8; ++e) {
      const int tok = b * TT + qt*64 + wave*16 + hf*8 + e;
      const int d = n*16 + l16;
      y[(size_t)tok * (HH*128) + h*128 + d] = (bf16)(oacc[n][e] / lrow[e]);
    }
}

// ---------------------------------------------------------------------------
__global__ __launch_bounds__(256)
void k_add(const float* __restrict__ a, const float* __restrict__ b,
           float* __restrict__ o, int n) {
  for (int i = blockIdx.x*256 + threadIdx.x; i < n; i += gridDim.x*256)
    o[i] = a[i] + b[i];
}

__global__ __launch_bounds__(256)
void k_silu_mul(const float* __restrict__ g1, const float* __restrict__ g3,
                bf16* __restrict__ o, int n) {
  for (int i = blockIdx.x*256 + threadIdx.x; i < n; i += gridDim.x*256) {
    const float g = g1[i];
    o[i] = (bf16)((g / (1.f + __expf(-g))) * g3[i]);
  }
}

// ---------------------------------------------------------------------------
extern "C" void kernel_launch(void* const* d_in, const int* in_sizes, int n_in,
                              void* d_out, int out_size, void* d_ws, size_t ws_size,
                              hipStream_t stream) {
  (void)in_sizes; (void)n_in; (void)out_size; (void)ws_size;
  const float* x          = (const float*)d_in[0];
  /* d_in[1] = mask (causal, synthesized in-kernel) */
  const float* fcos       = (const float*)d_in[2];
  const float* fsin       = (const float*)d_in[3];
  const float* attn_nw    = (const float*)d_in[4];
  const float* wq_a       = (const float*)d_in[5];
  const float* q_nw       = (const float*)d_in[6];
  const float* wq_b       = (const float*)d_in[7];
  const float* wkv_a      = (const float*)d_in[8];
  const float* kv_nw      = (const float*)d_in[9];
  const float* wkv_b      = (const float*)d_in[10];
  const float* wo         = (const float*)d_in[11];
  const float* ffn_nw     = (const float*)d_in[12];
  const float* w1         = (const float*)d_in[13];
  const float* w2         = (const float*)d_in[14];
  const float* w3         = (const float*)d_in[15];
  float* out = (float*)d_out;

  constexpr int Bb = 2, T = 1024, D = 2048, H = 16, NOPE = 128, ROPE = 64,
                QKD = 192, QLORA = 1536, KVLORA = 512, VHD = 128, INTER = 8192;
  const int M = Bb * T;  // 2048 tokens

  char* p = (char*)d_ws;
  auto alloc = [&](size_t bytes) -> char* {
    char* r = p;
    p += (bytes + 255) & ~(size_t)255;
    return r;
  };
  bf16*  h_bf    = (bf16*) alloc((size_t)M * D * 2);
  float* qlat    = (float*)alloc((size_t)M * QLORA * 4);
  bf16*  qlat_bf = (bf16*) alloc((size_t)M * QLORA * 2);
  float* qbuf    = (float*)alloc((size_t)M * H * QKD * 4);
  float* kvall   = (float*)alloc((size_t)M * (KVLORA + ROPE) * 4);
  bf16*  kvlatbf = (bf16*) alloc((size_t)M * KVLORA * 2);
  float* kvbuf   = (float*)alloc((size_t)M * H * (NOPE + VHD) * 4);
  bf16*  qfb     = (bf16*) alloc((size_t)M * H * QKD * 2);
  bf16*  kfb     = (bf16*) alloc((size_t)M * H * QKD * 2);
  bf16*  vtb     = (bf16*) alloc((size_t)Bb * H * VHD * T * 2);
  bf16*  yb      = (bf16*) alloc((size_t)M * H * VHD * 2);
  float* yo      = (float*)alloc((size_t)M * D * 4);
  float* x1      = (float*)alloc((size_t)M * D * 4);
  bf16*  h2      = (bf16*) alloc((size_t)M * D * 2);
  float* g1      = (float*)alloc((size_t)M * INTER * 4);
  float* g3      = (float*)alloc((size_t)M * INTER * 4);
  bf16*  gact    = (bf16*) alloc((size_t)M * INTER * 2);
  float* ffn     = (float*)alloc((size_t)M * D * 4);

  auto gemm = [&](const bf16* A, const float* Bw, float* C, int m, int n, int k) {
    dim3 grid((n + 127) / 128, m / 128);
    k_gemm<<<grid, 256, 0, stream>>>(A, Bw, C, m, n, k);
  };

  // attention path
  k_rmsnorm<<<M, 256, 0, stream>>>(x, attn_nw, h_bf, D, D);
  gemm(h_bf, wq_a, qlat, M, QLORA, D);
  k_rmsnorm<<<M, 256, 0, stream>>>(qlat, q_nw, qlat_bf, QLORA, QLORA);
  gemm(qlat_bf, wq_b, qbuf, M, H * QKD, QLORA);
  gemm(h_bf, wkv_a, kvall, M, KVLORA + ROPE, D);
  k_rmsnorm<<<M, 256, 0, stream>>>(kvall, kv_nw, kvlatbf, KVLORA, KVLORA + ROPE);
  gemm(kvlatbf, wkv_b, kvbuf, M, H * (NOPE + VHD), KVLORA);
  {
    const int total = M * H * 96;
    k_pack_qf<<<(total + 255) / 256, 256, 0, stream>>>(qbuf, fcos, fsin, qfb, total);
  }
  {
    const int total = M * H * 160;
    k_pack_kfv<<<(total + 255) / 256, 256, 0, stream>>>(kvbuf, kvall, fcos, fsin,
                                                        kfb, vtb, total);
  }
  k_attn<<<dim3(T / 64, H, Bb), 128, 0, stream>>>(qfb, kfb, vtb, yb);
  gemm(yb, wo, yo, M, D, H * VHD);
  k_add<<<4096, 256, 0, stream>>>(x, yo, x1, M * D);

  // FFN path
  k_rmsnorm<<<M, 256, 0, stream>>>(x1, ffn_nw, h2, D, D);
  gemm(h2, w1, g1, M, INTER, D);
  gemm(h2, w3, g3, M, INTER, D);
  k_silu_mul<<<8192, 256, 0, stream>>>(g1, g3, gact, M * INTER);
  gemm(gact, w2, ffn, M, D, INTER);
  k_add<<<4096, 256, 0, stream>>>(x1, ffn, out, M * D);
}
